// NonLocalBlock_26972394618943
// MI455X (gfx1250) — matrix operand order repaired and checked by reading, not compiled, not run on hardware
//
#include <hip/hip_runtime.h>
#include <stdint.h>

// ---------------------------------------------------------------------------
// NonLocalBlock (self-attention) for MI455X / gfx1250.
// B=4, C=128, H=W=64 -> N=4096.  All GEMMs via v_wmma_f32_16x16x32_bf16.
// Flash-attention: the 4x4096x4096 score matrix is never materialized.
// K/V chunks double-buffered in LDS via global_load_async_to_lds_b128
// (ASYNCcnt) so copy latency hides behind WMMA compute.
// ---------------------------------------------------------------------------

namespace {

constexpr int kC = 128;     // channels
constexpr int kN = 4096;    // H*W
// workspace layout, in bf16 (ushort) element units
constexpr size_t WT_OFF = 0;                 // w_theta bf16 [o][c]
constexpr size_t WP_OFF = 16384;             // w_phi
constexpr size_t WG_OFF = 32768;             // w_g
constexpr size_t WO_OFF = 49152;             // w_out
constexpr size_t THETA_OFF = 65536;          // theta^T  bf16 [b][n][c]
constexpr size_t BNC = (size_t)4 * kN * kC;  // 2097152
constexpr size_t PHI_OFF = THETA_OFF + BNC;  // phi^T    bf16 [b][n][c]
constexpr size_t G_OFF   = PHI_OFF + BNC;    // g        bf16 [b][c][n]
constexpr size_t OT_OFF  = G_OFF + BNC;      // O^T      bf16 [b][n][c]

typedef __attribute__((ext_vector_type(16))) __bf16 v16bf;
typedef __attribute__((ext_vector_type(8)))  float  v8f;

union Frag {
  uint4 q[2];
  v16bf v;
};

// hardware f32 -> bf16 convert (v_cvt_pk_bf16_f32)
__device__ __forceinline__ unsigned short f2bf(float f) {
  __bf16 h = (__bf16)f;
  return __builtin_bit_cast(unsigned short, h);
}

// async 16B global -> LDS copy, one per lane; tracked by ASYNCcnt.
// LDS byte address = low 32 bits of the generic pointer (flat-LDS aperture).
__device__ __forceinline__ void async_cp16(unsigned short* lds,
                                           const unsigned short* g) {
  unsigned l = (unsigned)(uintptr_t)lds;
  asm volatile("global_load_async_to_lds_b128 %0, %1, off"
               :: "v"(l), "v"(g) : "memory");
}
__device__ __forceinline__ void wait_async0() {
  asm volatile("s_wait_asynccnt 0x0" ::: "memory");
}

// A fragment (16x32 bf16, MxK). src pre-offset to (m_base,k_base), row-major,
// leading dim ld elements. Lane l holds row l&15; K halves 0-7/16-23 (lanes
// 0-15) and 8-15/24-31 (lanes 16-31).  Two 16B loads per lane.
__device__ __forceinline__ Frag load_a(const unsigned short* src, int ld, int lane) {
  int r = lane & 15;
  int h = (lane >> 4) << 3;  // 0 or 8
  Frag f;
  f.q[0] = *(const uint4*)(src + r * ld + h);
  f.q[1] = *(const uint4*)(src + r * ld + 16 + h);
  return f;
}

// B fragment (32x16 bf16, KxN). srcT is n-major: element(k,n) at srcT[n*ld+k],
// pre-offset to (k_base,n_base). Lane l holds column l&15; K halves 0-7/16-23
// (lanes 0-15) and 8-15/24-31 (lanes 16-31).  Two 16B loads per lane.
__device__ __forceinline__ Frag load_b(const unsigned short* srcT, int ld, int lane) {
  int n = lane & 15;
  int h = (lane >> 4) << 3;  // 0 or 8
  Frag f;
  f.q[0] = *(const uint4*)(srcT + n * ld + h);
  f.q[1] = *(const uint4*)(srcT + n * ld + 16 + h);
  return f;
}

__device__ __forceinline__ v8f wmma_bf16(const Frag& a, const Frag& b, v8f c) {
  return __builtin_amdgcn_wmma_f32_16x16x32_bf16(false, a.v, false, b.v,
                                                 (short)0, c, false, false);
}

}  // namespace

// ---------------------------------------------------------------------------
// Kernel 0: convert the four CxC weight matrices to bf16 in workspace.
// ---------------------------------------------------------------------------
__global__ __launch_bounds__(256) void wconv_kernel(
    const float* __restrict__ wt, const float* __restrict__ wp,
    const float* __restrict__ wg, const float* __restrict__ wo,
    unsigned short* __restrict__ ws) {
  int i = blockIdx.x * 256 + threadIdx.x;  // 0..16383
  ws[WT_OFF + i] = f2bf(wt[i]);
  ws[WP_OFF + i] = f2bf(wp[i]);
  ws[WG_OFF + i] = f2bf(wg[i]);
  ws[WO_OFF + i] = f2bf(wo[i]);
}

// ---------------------------------------------------------------------------
// Kernel 1: theta/phi/g 1x1 convs.  x tile staged (transposed) in LDS as bf16.
// theta^T, phi^T stored pixel-major [n][c]; g stored channel-major [c][n].
// grid: (N/128, B), block: 256 (8 waves); wave w owns output channels w*16..+16
// ---------------------------------------------------------------------------
__global__ __launch_bounds__(256) void qkv_kernel(
    const float* __restrict__ x, const float* __restrict__ bth,
    const float* __restrict__ bph, const float* __restrict__ bgv,
    unsigned short* __restrict__ ws) {
  constexpr int LDX = 136;  // padded LDS leading dim (16B-aligned rows)
  __shared__ alignas(16) unsigned short xT[128 * LDX];

  const int b = blockIdx.y;
  const int n0 = blockIdx.x * 128;
  const float* xb = x + (size_t)b * kC * kN;

  // cooperative load (float4) + transpose + f32->bf16 into LDS
#pragma unroll
  for (int it = 0; it < 16; ++it) {
    int idx = threadIdx.x + it * 256;  // float4 index, 0..4095
    int c = idx >> 5;                  // 32 float4 per channel row
    int n4 = (idx & 31) << 2;
    float4 vv = *(const float4*)(xb + (size_t)c * kN + n0 + n4);
    xT[(n4 + 0) * LDX + c] = f2bf(vv.x);
    xT[(n4 + 1) * LDX + c] = f2bf(vv.y);
    xT[(n4 + 2) * LDX + c] = f2bf(vv.z);
    xT[(n4 + 3) * LDX + c] = f2bf(vv.w);
  }
  __syncthreads();

  const int lane = threadIdx.x & 31;
  const int wave = threadIdx.x >> 5;
  const int col = lane & 15;
  const int hh = lane >> 4;
  const int o0 = wave * 16;

  unsigned short* theta = ws + THETA_OFF + (size_t)b * kN * kC;
  unsigned short* phi   = ws + PHI_OFF   + (size_t)b * kN * kC;
  unsigned short* gm    = ws + G_OFF     + (size_t)b * kC * kN;

  const float bth_c = bth[o0 + col];
  const float bph_c = bph[o0 + col];
  float bg_r[8];
#pragma unroll
  for (int v = 0; v < 8; ++v) bg_r[v] = bgv[o0 + hh * 8 + v];

  // per-wave invariant weight fragments
  Frag wtB[4], wpB[4], wgA[4];
#pragma unroll
  for (int ks = 0; ks < 4; ++ks) {
    wtB[ks] = load_b(ws + WT_OFF + (size_t)o0 * kC + ks * 32, kC, lane);
    wpB[ks] = load_b(ws + WP_OFF + (size_t)o0 * kC + ks * 32, kC, lane);
    wgA[ks] = load_a(ws + WG_OFF + (size_t)o0 * kC + ks * 32, kC, lane);
  }

  for (int pt = 0; pt < 8; ++pt) {
    v8f dth = {}, dph = {}, dg = {};
#pragma unroll
    for (int ks = 0; ks < 4; ++ks) {
      Frag xa = load_a(xT + pt * 16 * LDX + ks * 32, LDX, lane);
      dth = wmma_bf16(xa, wtB[ks], dth);
      dph = wmma_bf16(xa, wpB[ks], dph);
      Frag xbf = load_b(xT + pt * 16 * LDX + ks * 32, LDX, lane);
      dg = wmma_bf16(wgA[ks], xbf, dg);
    }
    // theta/phi tile: M=pixel, N=channel -> [n][c] stores (contiguous per VGPR)
#pragma unroll
    for (int v = 0; v < 8; ++v) {
      int p = n0 + pt * 16 + hh * 8 + v;
      theta[(size_t)p * kC + o0 + col] = f2bf(dth[v] + bth_c);
      phi[(size_t)p * kC + o0 + col]   = f2bf(dph[v] + bph_c);
    }
    // g tile: M=channel, N=pixel -> [c][n] stores
#pragma unroll
    for (int v = 0; v < 8; ++v) {
      int oc = o0 + hh * 8 + v;
      gm[(size_t)oc * kN + n0 + pt * 16 + col] = f2bf(dg[v] + bg_r[v]);
    }
  }
}

// ---------------------------------------------------------------------------
// Kernel 2: flash attention.  Each wave: 16 queries, full C=128 accumulator,
// streams 4096 keys in chunks of 64.  phi/g chunks double-buffered in LDS via
// async copies; score matrix never hits memory.
// grid: (N/128, B), block 256 (8 waves).
// ---------------------------------------------------------------------------
__global__ __launch_bounds__(256) void attn_kernel(unsigned short* __restrict__ ws) {
  constexpr int LDK = 136;  // sPhi leading dim (keys x C)
  constexpr int LDG = 72;   // sG leading dim (C x keys), also P-tile ld
  __shared__ alignas(16) unsigned short sPhi[2][64 * LDK];    // 2 x 17.0 KB
  __shared__ alignas(16) unsigned short sG[2][128 * LDG];     // 2 x 18.0 KB
  __shared__ alignas(16) unsigned short Pbuf[8 * 16 * LDG];   // 18.0 KB

  const int b = blockIdx.y;
  const int lane = threadIdx.x & 31;
  const int wave = threadIdx.x >> 5;
  const int col = lane & 15;
  const int hh = lane >> 4;
  const int q0 = (blockIdx.x * 8 + wave) * 16;

  const unsigned short* theta = ws + THETA_OFF + (size_t)b * kN * kC;
  const unsigned short* phi   = ws + PHI_OFF   + (size_t)b * kN * kC;
  const unsigned short* gm    = ws + G_OFF     + (size_t)b * kC * kN;
  unsigned short* ot          = ws + OT_OFF    + (size_t)b * kN * kC;

  // issue async stage of chunk (8 x 16B per thread): phi [64 k][128 c],
  // g [128 c][64 k]
  auto stage = [&](int buf, int kb) {
#pragma unroll
    for (int it = 0; it < 4; ++it) {
      int idx = threadIdx.x + it * 256;  // uint4 index, 0..1023
      int row = idx >> 4;                // 16 uint4 per 128-elem phi row
      int c8 = (idx & 15) << 3;
      async_cp16(&sPhi[buf][row * LDK + c8],
                 phi + (size_t)(kb + row) * kC + c8);
    }
#pragma unroll
    for (int it = 0; it < 4; ++it) {
      int idx = threadIdx.x + it * 256;
      int row = idx >> 3;                // 8 uint4 per 64-elem g row
      int k8 = (idx & 7) << 3;
      async_cp16(&sG[buf][row * LDG + k8],
                 gm + (size_t)row * kN + kb + k8);
    }
  };

  stage(0, 0);  // prologue: chunk 0 in flight

  // query A-fragments (live in registers for whole kernel)
  Frag qa[4];
#pragma unroll
  for (int ks = 0; ks < 4; ++ks)
    qa[ks] = load_a(theta + (size_t)q0 * kC + ks * 32, kC, lane);

  v8f O[8];
#pragma unroll
  for (int ct = 0; ct < 8; ++ct) O[ct] = {};
  float mrow[8], lrow[8];
#pragma unroll
  for (int v = 0; v < 8; ++v) { mrow[v] = -3.0e38f; lrow[v] = 0.f; }

  unsigned short* myP = Pbuf + wave * 16 * LDG;

  for (int i = 0; i < kN / 64; ++i) {
    const int kb = i * 64;
    const int buf = i & 1;
    wait_async0();    // this wave's chunk-i copies have landed
    __syncthreads();  // -> every wave's copies have landed; prev compute done
    if (i + 1 < kN / 64) stage(buf ^ 1, kb + 64);  // overlap with compute

    // ---- S = theta^T phi  (16 queries x 64 keys), K = C = 128
    v8f s[4];
#pragma unroll
    for (int t = 0; t < 4; ++t) {
      v8f acc = {};
#pragma unroll
      for (int ks = 0; ks < 4; ++ks) {
        Frag kf = load_b(&sPhi[buf][(t * 16) * LDK + ks * 32], LDK, lane);
        acc = wmma_bf16(qa[ks], kf, acc);
      }
      s[t] = acc;
    }
    // ---- online softmax (row reductions within each 16-lane half)
    float alpha[8];
#pragma unroll
    for (int v = 0; v < 8; ++v) {
      float rm = fmaxf(fmaxf(s[0][v], s[1][v]), fmaxf(s[2][v], s[3][v]));
#pragma unroll
      for (int off = 1; off < 16; off <<= 1)
        rm = fmaxf(rm, __shfl_xor(rm, off, 32));
      float mn = fmaxf(mrow[v], rm);
      alpha[v] = __expf(mrow[v] - mn);
      mrow[v] = mn;
    }
#pragma unroll
    for (int v = 0; v < 8; ++v) {
      float rs = 0.f;
#pragma unroll
      for (int t = 0; t < 4; ++t) {
        float p = __expf(s[t][v] - mrow[v]);
        s[t][v] = p;
        rs += p;
      }
#pragma unroll
      for (int off = 1; off < 16; off <<= 1) rs += __shfl_xor(rs, off, 32);
      lrow[v] = lrow[v] * alpha[v] + rs;
    }
#pragma unroll
    for (int ct = 0; ct < 8; ++ct)
#pragma unroll
      for (int v = 0; v < 8; ++v) O[ct][v] *= alpha[v];

    // ---- P (C/D layout) -> per-wave LDS scratch -> reload as A-fragments
#pragma unroll
    for (int t = 0; t < 4; ++t)
#pragma unroll
      for (int v = 0; v < 8; ++v)
        myP[(hh * 8 + v) * LDG + t * 16 + col] = f2bf(s[t][v]);
    asm volatile("s_wait_dscnt 0x0" ::: "memory");
    __builtin_amdgcn_wave_barrier();

    // ---- O += P * g   (K = 64 keys, N = 128 channels)
#pragma unroll
    for (int ks2 = 0; ks2 < 2; ++ks2) {
      Frag pa = load_a(myP + ks2 * 32, LDG, lane);
#pragma unroll
      for (int ct = 0; ct < 8; ++ct) {
        Frag gf = load_b(&sG[buf][(ct * 16) * LDG + ks2 * 32], LDG, lane);
        O[ct] = wmma_bf16(pa, gf, O[ct]);
      }
    }
  }

  // ---- finalize: O^T [n][c] in bf16
  float inv[8];
#pragma unroll
  for (int v = 0; v < 8; ++v) inv[v] = 1.0f / lrow[v];
#pragma unroll
  for (int ct = 0; ct < 8; ++ct)
#pragma unroll
    for (int v = 0; v < 8; ++v) {
      int q = q0 + hh * 8 + v;
      ot[(size_t)q * kC + ct * 16 + col] = f2bf(O[ct][v] * inv[v]);
    }
}

// ---------------------------------------------------------------------------
// Kernel 3: out = w_out * O + b_out + x, stored [b][c][n] f32 (coalesced).
// O^T pixel-block staged once in LDS via async copies (shared by all waves).
// grid: (N/128, B), block 256 (8 waves); wave w owns channels w*16..+16
// ---------------------------------------------------------------------------
__global__ __launch_bounds__(256) void out_kernel(
    const float* __restrict__ x, const float* __restrict__ bo,
    const unsigned short* __restrict__ ws, float* __restrict__ out) {
  constexpr int LDO = 136;
  __shared__ alignas(16) unsigned short sOT[128 * LDO];  // 34 KB

  const int b = blockIdx.y;
  const int n0 = blockIdx.x * 128;
  const int lane = threadIdx.x & 31;
  const int wave = threadIdx.x >> 5;
  const int col = lane & 15;
  const int hh = lane >> 4;
  const int o0 = wave * 16;

  const unsigned short* ot = ws + OT_OFF + (size_t)b * kN * kC;
  const float* xb = x + (size_t)b * kC * kN;
  float* ob = out + (size_t)b * kC * kN;

  // async-stage O^T block [128 pix][128 c] into LDS (2048 x 16B)
#pragma unroll
  for (int it = 0; it < 8; ++it) {
    int idx = threadIdx.x + it * 256;  // uint4 index, 0..2047
    int row = idx >> 4;                // 16 uint4 per 128-elem row
    int c8 = (idx & 15) << 3;
    async_cp16(&sOT[row * LDO + c8], ot + (size_t)(n0 + row) * kC + c8);
  }

  Frag wa[4];
#pragma unroll
  for (int ks = 0; ks < 4; ++ks)
    wa[ks] = load_a(ws + WO_OFF + (size_t)o0 * kC + ks * 32, kC, lane);
  float br[8];
#pragma unroll
  for (int v = 0; v < 8; ++v) br[v] = bo[o0 + hh * 8 + v];

  wait_async0();
  __syncthreads();

  for (int pt = 0; pt < 8; ++pt) {
    v8f d = {};
#pragma unroll
    for (int ks = 0; ks < 4; ++ks) {
      Frag of = load_b(&sOT[(pt * 16) * LDO + ks * 32], LDO, lane);
      d = wmma_bf16(wa[ks], of, d);
    }
#pragma unroll
    for (int v = 0; v < 8; ++v) {
      int oc = o0 + hh * 8 + v;
      size_t idx = (size_t)oc * kN + n0 + pt * 16 + col;
      ob[idx] = d[v] + br[v] + xb[idx];
    }
  }
}

// ---------------------------------------------------------------------------
extern "C" void kernel_launch(void* const* d_in, const int* in_sizes, int n_in,
                              void* d_out, int out_size, void* d_ws, size_t ws_size,
                              hipStream_t stream) {
  const float* x       = (const float*)d_in[0];
  const float* w_theta = (const float*)d_in[1];
  const float* b_theta = (const float*)d_in[2];
  const float* w_phi   = (const float*)d_in[3];
  const float* b_phi   = (const float*)d_in[4];
  const float* w_g     = (const float*)d_in[5];
  const float* b_g     = (const float*)d_in[6];
  const float* w_out   = (const float*)d_in[7];
  const float* b_out   = (const float*)d_in[8];
  unsigned short* ws = (unsigned short*)d_ws;
  float* out = (float*)d_out;

  wconv_kernel<<<64, 256, 0, stream>>>(w_theta, w_phi, w_g, w_out, ws);
  qkv_kernel<<<dim3(kN / 128, 4), 256, 0, stream>>>(x, b_theta, b_phi, b_g, ws);
  attn_kernel<<<dim3(kN / 128, 4), 256, 0, stream>>>(ws);
  out_kernel<<<dim3(kN / 128, 4), 256, 0, stream>>>(x, b_out, ws, out);
}
